// MPSClassifier_68212670595937
// MI455X (gfx1250) — matrix operand order repaired and checked by reading, not compiled, not run on hardware
//
#include <hip/hip_runtime.h>

typedef __attribute__((ext_vector_type(16))) _Float16 v16h;
typedef __attribute__((ext_vector_type(8)))  _Float16 v8h;
typedef __attribute__((ext_vector_type(2)))  _Float16 h2;
typedef __attribute__((ext_vector_type(8)))  float    v8f;
typedef __attribute__((ext_vector_type(4)))  unsigned int u32x4;
typedef __attribute__((ext_vector_type(8)))  int i32x8;
typedef __attribute__((ext_vector_type(4)))  int i32x4;

#define NSITES 784
#define NMID   782
#define BATCH  8192
#define NCLS   10
#define PIH    1.5707963267948966f
#define EPS    1e-8f

// ---------------- prep: transpose mid cores to f16  At[s][f][k'][j] = A[s][f][j][k'] -------------
__global__ __launch_bounds__(256) void mps_prep_mid(const float* __restrict__ cm,
                                                    _Float16* __restrict__ at) {
    int e = blockIdx.x * 256 + threadIdx.x;
    if (e >= NMID * 2 * 32 * 32) return;
    int j  = e & 31;
    int kp = (e >> 5) & 31;
    int sf = e >> 10;                       // s*2 + f
    at[e] = (_Float16)cm[(size_t)sf * 1024 + j * 32 + kp];
}

// prep: transposed, zero-padded last core  CLt[f][c][j] = CL[f][j][c], c in [0,16)
__global__ __launch_bounds__(256) void mps_prep_last(const float* __restrict__ cl,
                                                     _Float16* __restrict__ clt) {
    int e = blockIdx.x * 256 + threadIdx.x;
    if (e >= 2 * 16 * 32) return;
    int j = e & 31;
    int c = (e >> 5) & 15;
    int f = e >> 9;
    clt[e] = (c < NCLS) ? (_Float16)cl[(f * 32 + j) * NCLS + c] : (_Float16)0.0f;
}

// load WMMA A-operand slice for this lane from a row-major [*,32]-f16 row.
// A k order chosen here (16x32, not the ISA table order): lanes 0-15 -> K 0..15 ; lanes 16-31 -> K 16..31 (contiguous, matches B)
__device__ __forceinline__ v16h load_a(const _Float16* row, int hi) {
    v8h a = *(const v8h*)(row + 16 * hi);
    v8h b = *(const v8h*)(row + 16 * hi + 8);
    v16h r;
#pragma unroll
    for (int i = 0; i < 8; ++i) { r[i] = a[i]; r[8 + i] = b[i]; }
    return r;
}

__device__ __forceinline__ v8f wmma_f16(v16h a, v16h b, v8f c) {
    return __builtin_amdgcn_wmma_f32_16x16x32_f16(false, a, false, b, (short)0, c, false, false);
}

__device__ __forceinline__ h2 pk_h2(float a, float b) {
    return __builtin_bit_cast(h2, __builtin_amdgcn_cvt_pkrtz(a, b));
}

// ---- Tensor Data Mover: async 4KB (2048 x f16) 1-D tile, global -> LDS (D# per ISA 08 §8.3/8.4)
__device__ __forceinline__ void tdm_load_core(const _Float16* gsrc, unsigned lds_off) {
    unsigned long long ga = (unsigned long long)(uintptr_t)gsrc;
    u32x4 g0;
    g0[0] = 1u;                                                  // count=1 user descriptor
    g0[1] = lds_off;                                             // lds_addr (bytes)
    g0[2] = (unsigned)ga;                                        // global_addr[31:0]
    g0[3] = (unsigned)((ga >> 32) & 0x01FFFFFFu) | (2u << 30);   // global_addr[56:32] | type=2
    i32x8 g1;
    g1[0] = 0x00010000;   // workgroup_mask=0, data_size=1 (2 bytes)
    g1[1] = 0x08000000;   // tensor_dim0 = 2048  (bits 79:48)
    g1[2] = 0x00010000;   // tensor_dim1 = 1     (bits 111:80)
    g1[3] = 0x08000000;   // tile_dim0  = 2048   (bits 127:112)
    g1[4] = 0;            // tile_dim1 = tile_dim2 = 0 (unused)
    g1[5] = 2048;         // tensor_dim0_stride  (bits 207:160)
    g1[6] = 0;
    g1[7] = 0;
    i32x4 z4 = {0, 0, 0, 0};
#if __clang_major__ >= 23
    i32x8 z8 = {};
    __builtin_amdgcn_tensor_load_to_lds(g0, g1, z4, z4, z8, 0);
#else
    __builtin_amdgcn_tensor_load_to_lds(g0, g1, z4, z4, 0);
#endif
}

// ------------- main chain kernel: ONE wave per workgroup, 16 batch rows, all 784 sites ----------
__global__ __launch_bounds__(32) void mps_chain(const float* __restrict__ x,
                                                const float* __restrict__ core_first,
                                                const _Float16* __restrict__ at_mid,
                                                const _Float16* __restrict__ clt,
                                                float* __restrict__ out) {
    __shared__ __align__(16) _Float16 ldsbuf[4][2048];      // quad buffer, 16KB (wave-private)

    const int lane = threadIdx.x & 31;
    const int m    = lane & 15;            // batch column within tile / A row
    const int hi   = lane >> 4;            // half-wave id
    const int b    = blockIdx.x * 16 + m;

    const float* __restrict__ xrow = x + (size_t)b * NSITES;

    // ---- site 0: M[k,b] = cos0*CF0[k] + sin0*CF1[k], stored raw in B-operand layout
    float a0 = PIH * xrow[0];
    float c0 = __cosf(a0), s0 = __sinf(a0);
    h2 Mh[8];
#pragma unroll
    for (int i = 0; i < 8; ++i) {
        int k0 = hi * 16 + 2 * i;
        float e0 = c0 * core_first[k0]     + s0 * core_first[32 + k0];
        float e1 = c0 * core_first[k0 + 1] + s0 * core_first[32 + k0 + 1];
        Mh[i] = pk_h2(e0, e1);
    }

    const v8f zero = {};
    float inv = 1.0f;                        // deferred 1/(norm+eps); first M is unnormalized

    // prologue: stage cores for steps 0..2 (prefetch depth 3) + first angle
    tdm_load_core(at_mid,        (unsigned)(uintptr_t)&ldsbuf[0][0]);
    tdm_load_core(at_mid + 2048, (unsigned)(uintptr_t)&ldsbuf[1][0]);
    tdm_load_core(at_mid + 4096, (unsigned)(uintptr_t)&ldsbuf[2][0]);
    float xa   = PIH * xrow[1];
    float craw = __cosf(xa), sraw = __sinf(xa);

    // ---- 782 middle sites
    for (int s = 0; s < NMID; ++s) {
        // previous step's normalization folded into this step's per-lane scalars
        float cv = craw * inv;
        float sv = sraw * inv;
        h2 ch = pk_h2(cv, cv);
        h2 sh = pk_h2(sv, sv);

        v16h B0, B1;
#pragma unroll
        for (int i = 0; i < 8; ++i) {
            h2 p0 = Mh[i] * ch;
            h2 p1 = Mh[i] * sh;
            B0[2 * i] = p0.x; B0[2 * i + 1] = p0.y;
            B1[2 * i] = p1.x; B1[2 * i + 1] = p1.y;
        }

        // stage the core 3 steps ahead (quad buffer, no barriers)
        int sn = (s + 3 < NMID) ? (s + 3) : (NMID - 1);
        tdm_load_core(at_mid + (size_t)sn * 2048,
                      (unsigned)(uintptr_t)&ldsbuf[(s + 3) & 3][0]);
        __builtin_amdgcn_s_wait_tensorcnt(3);   // oldest in-flight TDM (step s) complete
        asm volatile("" ::: "memory");

        const _Float16* ab = &ldsbuf[s & 3][0];           // [f][k'][j]
        v16h A0U = load_a(ab +            m * 32, hi);    // f=0, rows 0..15
        v16h A1U = load_a(ab + 1024 +     m * 32, hi);    // f=1, rows 0..15
        v16h A0L = load_a(ab + (16 + m) * 32, hi);        // f=0, rows 16..31
        v16h A1L = load_a(ab + 1024 + (16 + m) * 32, hi); // f=1, rows 16..31

        // four independent WMMAs (no D->C chaining hazards)
        v8f tU0 = wmma_f16(A0U, B0, zero);
        v8f tU1 = wmma_f16(A1U, B1, zero);
        v8f tL0 = wmma_f16(A0L, B0, zero);
        v8f tL1 = wmma_f16(A1L, B1, zero);

        // software-pipelined next-step transcendentals: independent work that fills the
        // WMMA->VALU hazard window (s+2 <= 783 always; s=781 yields the last-site angle)
        float xn = PIH * xrow[s + 2];
        craw = __cosf(xn);
        sraw = __sinf(xn);

        v8f tU = tU0 + tU1;
        v8f tL = tL0 + tL1;

        // per-batch-column Frobenius norm: in-lane 16 squares + cross-half add
        float ss = 0.f;
#pragma unroll
        for (int i = 0; i < 8; ++i) ss += tU[i] * tU[i] + tL[i] * tL[i];
        ss += __shfl_xor(ss, 16, 32);
        inv = __builtin_amdgcn_rcpf(__builtin_amdgcn_sqrtf(ss) + EPS);  // consumed next iter

        // D-layout -> B-layout: pack pairs, single exchange per pair (send what partner needs)
#pragma unroll
        for (int i = 0; i < 4; ++i) {
            int iU = __builtin_bit_cast(int, pk_h2(tU[2 * i], tU[2 * i + 1]));
            int iL = __builtin_bit_cast(int, pk_h2(tL[2 * i], tL[2 * i + 1]));
            int send = hi ? iU : iL;            // partner's missing half
            int recv = __shfl_xor(send, 16, 32);
            Mh[i]     = __builtin_bit_cast(h2, hi ? recv : iU);  // K 2i..2i+1  (+16 for hi half)
            Mh[4 + i] = __builtin_bit_cast(h2, hi ? iL : recv);  // K 8+2i..    (+16 for hi half)
        }
    }

    // ---- last site: logits = CLt0*(cos M) + CLt1*(sin M); craw/sraw already hold site-783 vals
    float cl = craw * inv;                  // fold last normalization
    float sl = sraw * inv;
    h2 chl = pk_h2(cl, cl);
    h2 shl = pk_h2(sl, sl);
    v16h B0, B1;
#pragma unroll
    for (int i = 0; i < 8; ++i) {
        h2 p0 = Mh[i] * chl;
        h2 p1 = Mh[i] * shl;
        B0[2 * i] = p0.x; B0[2 * i + 1] = p0.y;
        B1[2 * i] = p1.x; B1[2 * i + 1] = p1.y;
    }
    v16h C0 = load_a(clt +       m * 32, hi);
    v16h C1 = load_a(clt + 512 + m * 32, hi);
    v8f acc0 = wmma_f16(C0, B0, zero);
    v8f acc1 = wmma_f16(C1, B1, zero);
    v8f acc = acc0 + acc1;

#pragma unroll
    for (int r = 0; r < 8; ++r) {
        int c = r + 8 * hi;
        if (c < NCLS) out[(size_t)b * NCLS + c] = acc[r];
    }
}

extern "C" void kernel_launch(void* const* d_in, const int* in_sizes, int n_in,
                              void* d_out, int out_size, void* d_ws, size_t ws_size,
                              hipStream_t stream) {
    const float* x  = (const float*)d_in[0];   // [8192, 784]
    const float* cf = (const float*)d_in[1];   // [2, 1, 32]
    const float* cm = (const float*)d_in[2];   // [782, 2, 32, 32]
    const float* cl = (const float*)d_in[3];   // [2, 32, 10]
    float* out = (float*)d_out;

    _Float16* at_mid = (_Float16*)d_ws;                                    // 782*2*32*32 f16
    _Float16* clt    = (_Float16*)((char*)d_ws + (size_t)NMID * 2048 * 2); // 2*16*32 f16

    int nmid_el = NMID * 2 * 32 * 32;
    mps_prep_mid<<<(nmid_el + 255) / 256, 256, 0, stream>>>(cm, at_mid);
    mps_prep_last<<<(1024 + 255) / 256, 256, 0, stream>>>(cl, clt);

    // 512 independent one-wave workgroups (16 batch rows each) -> scheduler spreads over all WGPs
    mps_chain<<<BATCH / 16, 32, 0, stream>>>(x, cf, at_mid, clt, out);
}
